// TransformerLayer_50964081934911
// MI455X (gfx1250) — matrix operand order repaired and checked by reading, not compiled, not run on hardware
//
#include <hip/hip_runtime.h>
#include <hip/hip_bf16.h>

// ---------------- constants (match reference) ----------------
#define kS    4096
#define kD    1024
#define kDFF  4096
#define kH    16
#define kDK   64
#define LN_EPS 1e-5f

// ---------------- WMMA types ----------------
typedef _Float16 v16h __attribute__((ext_vector_type(16)));
typedef float    v8f  __attribute__((ext_vector_type(8)));

union F4H8 { float4 f4; _Float16 h[8]; };
union FragU { float4 f4[2]; v16h v; };

// ---------------- CDNA5 async global->LDS copy (ASYNCcnt) ----------------
// Each lane copies 16B from its global address directly into LDS, bypassing VGPRs.
__device__ __forceinline__ void async_b128(unsigned lds_addr, const void* gaddr) {
    asm volatile("global_load_async_to_lds_b128 %0, %1, off"
                 :: "v"(lds_addr), "v"(gaddr) : "memory");
}
__device__ __forceinline__ void wait_async0() {
    asm volatile("s_wait_asynccnt 0x0" ::: "memory");
}
// Low 32 bits of a generic LDS pointer are the LDS byte address (ISA aperture rule).
__device__ __forceinline__ unsigned lds_addr32(const void* p) {
    return (unsigned)(unsigned long long)p;
}

// Load A-operand fragment (16x32 f16) from LDS, row-major tile with stride lda.
// ISA 7.12.2 (16-bit A 16x32): lane L holds row M=L%16;
// halves 0..7 = K ko..ko+7, halves 8..15 = K ko+16..ko+23, ko=(L/16)*8.
__device__ __forceinline__ v16h load_frag_a(const _Float16* base, int lda, int lane) {
    int row = lane & 15;
    int ko  = (lane >> 4) << 3;
    const _Float16* p = base + row * lda + ko;
    FragU u;
    u.f4[0] = *(const float4*)(p);
    u.f4[1] = *(const float4*)(p + 16);
    return u.v;
}

// Load B-operand fragment (32x16 f16) from LDS stored as B^T [n][k], stride ldb.
// Lane L holds column N=L%16; halves 0..7 = K ko..ko+7, halves 8..15 = K ko+16..ko+23, ko=(L/16)*8.
__device__ __forceinline__ v16h load_frag_b(const _Float16* baseT, int ldb, int lane) {
    int n  = lane & 15;
    int ko = (lane >> 4) << 3;
    const _Float16* p = baseT + n * ldb + ko;
    FragU u;
    u.f4[0] = *(const float4*)(p);
    u.f4[1] = *(const float4*)(p + 16);
    return u.v;
}

// ---------------- cast helpers ----------------
__global__ void cast_f32_to_f16(const float* __restrict__ in,
                                _Float16* __restrict__ out, int n) {
    for (int i = blockIdx.x * blockDim.x + threadIdx.x; i < n; i += gridDim.x * blockDim.x)
        out[i] = (_Float16)in[i];
}

// w[K][N] fp32 (row-major) -> wT[N][K] f16
__global__ __launch_bounds__(256) void transpose_cast_f16(const float* __restrict__ w,
                                                          _Float16* __restrict__ wT,
                                                          int K, int N) {
    __shared__ float tile[32][33];
    int k0 = blockIdx.y * 32, n0 = blockIdx.x * 32;
    int tx = threadIdx.x, ty = threadIdx.y;      // 32 x 8
    for (int i = 0; i < 32; i += 8)
        tile[ty + i][tx] = w[(size_t)(k0 + ty + i) * N + n0 + tx];
    __syncthreads();
    for (int i = 0; i < 32; i += 8)
        wT[(size_t)(n0 + ty + i) * K + k0 + tx] = (_Float16)tile[tx][ty + i];
}

// ---------------- WMMA GEMM: C[M,N] = A[M,K] * B[K,N] + bias ----------------
// A row-major f16, BT = B^T [N][K] f16. Optional exact GELU. f32 and/or f16 stores.
// Block 256 threads = 8 waves (2x4). Tile 128x128, K-step 64, double-buffered LDS
// filled with global_load_async_to_lds_b128.
__global__ __launch_bounds__(256) void gemm_f16_wmma(const _Float16* __restrict__ A,
                                                     const _Float16* __restrict__ BT,
                                                     const float* __restrict__ bias,
                                                     float* __restrict__ Cf,
                                                     _Float16* __restrict__ Ch,
                                                     int M, int N, int K, int do_gelu) {
    const int LDT = 72;                   // 64 halves + 8 pad (144B rows, 16B aligned)
    const int TILEH = 128 * LDT;          // halves per tile buffer
    __shared__ __align__(16) _Float16 sA[2 * TILEH];
    __shared__ __align__(16) _Float16 sB[2 * TILEH];

    int tid = threadIdx.x, lane = tid & 31, wave = tid >> 5;
    int wm = wave >> 2;                   // 0..1 -> 64 rows
    int wn = wave & 3;                    // 0..3 -> 32 cols
    int m0 = blockIdx.y * 128, n0 = blockIdx.x * 128;

    // staging geometry: 1024 b128 units per 128x64 tile -> 4 per thread
    int r0 = tid >> 3, c0 = (tid & 7) << 3;     // rows r0 + 32j, fixed column
    const _Float16* gA = A + (size_t)(m0 + r0) * K + c0;
    const _Float16* gB = BT + (size_t)(n0 + r0) * K + c0;
    unsigned lA0 = lds_addr32(&sA[r0 * LDT + c0]);
    unsigned lB0 = lds_addr32(&sB[r0 * LDT + c0]);
    const unsigned BUFB = (unsigned)(TILEH * sizeof(_Float16));
    const unsigned ROWJ = (unsigned)(32u * LDT * sizeof(_Float16));

    v8f acc[4][2];
    for (int i = 0; i < 4; i++)
        for (int j = 0; j < 2; j++)
            for (int e = 0; e < 8; e++) acc[i][j][e] = 0.0f;

    const int ksteps = K >> 6;
    // preload tile 0 -> buffer 0
    for (int j = 0; j < 4; j++) {
        async_b128(lA0 + (unsigned)j * ROWJ, gA + (size_t)(j * 32) * K);
        async_b128(lB0 + (unsigned)j * ROWJ, gB + (size_t)(j * 32) * K);
    }
    wait_async0();
    __syncthreads();

    int buf = 0;
    for (int kt = 0; kt < ksteps; kt++) {
        if (kt + 1 < ksteps) {            // issue async copies for next tile
            const _Float16* nA = gA + (size_t)(kt + 1) * 64;
            const _Float16* nB = gB + (size_t)(kt + 1) * 64;
            unsigned nb = (unsigned)(buf ^ 1) * BUFB;
            for (int j = 0; j < 4; j++) {
                async_b128(lA0 + nb + (unsigned)j * ROWJ, nA + (size_t)(j * 32) * K);
                async_b128(lB0 + nb + (unsigned)j * ROWJ, nB + (size_t)(j * 32) * K);
            }
        }
        const _Float16* bA = sA + buf * TILEH;
        const _Float16* bB = sB + buf * TILEH;
        for (int ks = 0; ks < 2; ks++) {
            v16h bf0 = load_frag_b(&bB[(wn * 32 +  0) * LDT + ks * 32], LDT, lane);
            v16h bf1 = load_frag_b(&bB[(wn * 32 + 16) * LDT + ks * 32], LDT, lane);
            for (int i = 0; i < 4; i++) {
                v16h af = load_frag_a(&bA[(wm * 64 + i * 16) * LDT + ks * 32], LDT, lane);
                acc[i][0] = __builtin_amdgcn_wmma_f32_16x16x32_f16(false, af, false, bf0,
                                                                   (short)0, acc[i][0], false, false);
                acc[i][1] = __builtin_amdgcn_wmma_f32_16x16x32_f16(false, af, false, bf1,
                                                                   (short)0, acc[i][1], false, false);
            }
        }
        wait_async0();
        __syncthreads();
        buf ^= 1;
    }

    // epilogue; C layout: row m = v + 8*(lane/16), col n = lane%16
    int mlo = (lane >> 4) << 3, nn = lane & 15;
    for (int i = 0; i < 4; i++)
        for (int j = 0; j < 2; j++) {
            int col = n0 + wn * 32 + j * 16 + nn;
            float bv = bias ? bias[col] : 0.0f;
            for (int v = 0; v < 8; v++) {
                int row = m0 + wm * 64 + i * 16 + mlo + v;
                float val = acc[i][j][v] + bv;
                if (do_gelu) val = 0.5f * val * (1.0f + erff(val * 0.70710678118654752f));
                size_t idx = (size_t)row * N + col;
                if (Cf) Cf[idx] = val;
                if (Ch) Ch[idx] = (_Float16)val;
            }
        }
}

// ---------------- Flash attention ----------------
// grid (S/128, H), block 256 (8 waves); each wave owns 16 query rows.
// 128-query tiles halve K/V traffic vs 64; sK/sVt double-buffered: K(t+1) via
// async-LDS copies and V(t+1) via registers, both issued before the tile-t math.
__global__ __launch_bounds__(256) void flash_attn_wmma(const _Float16* __restrict__ Q,
                                                       const _Float16* __restrict__ Kv,
                                                       const _Float16* __restrict__ V,
                                                       _Float16* __restrict__ ctx) {
    const int LD = 72;                    // 64 halves + 8 pad
    const int KVH = 64 * LD;              // halves per K/V buffer
    __shared__ __align__(16) _Float16 sQ[128 * LD];
    __shared__ __align__(16) _Float16 sK[2 * KVH];
    __shared__ __align__(16) _Float16 sVt[2 * KVH];   // V^T: [dim][key]
    __shared__ __align__(16) _Float16 sP[8 * 16 * LD];

    int tid = threadIdx.x, lane = tid & 31, wave = tid >> 5;
    int head = blockIdx.y;
    int q0 = blockIdx.x * 128;
    int colBase = head * kDK;

    // staging geometry: 64x64 tile = 512 b128 units -> 2/thread (rows r0, r0+32)
    //                   128x64 Q tile = 1024 units -> 4/thread (rows r0+32j)
    int r0 = tid >> 3, c0 = (tid & 7) << 3;
    const _Float16* gQ = Q + (size_t)(q0 + r0) * kD + colBase + c0;
    const _Float16* gK = Kv + (size_t)r0 * kD + colBase + c0;
    const _Float16* gV = V + (size_t)r0 * kD + colBase + c0;
    unsigned lQ0 = lds_addr32(&sQ[r0 * LD + c0]);
    unsigned lK0 = lds_addr32(&sK[r0 * LD + c0]);
    const unsigned ROWJ32 = (unsigned)(32u * LD * sizeof(_Float16));
    const unsigned KBUFB  = (unsigned)(KVH * sizeof(_Float16));

    // prologue: async Q (whole 128x64) + async K tile 0; V tile 0 through regs
    for (int j = 0; j < 4; j++)
        async_b128(lQ0 + (unsigned)j * ROWJ32, gQ + (size_t)(j * 32) * kD);
    for (int j = 0; j < 2; j++)
        async_b128(lK0 + (unsigned)j * ROWJ32, gK + (size_t)(j * 32) * kD);
    F4H8 vreg[2];
    for (int j = 0; j < 2; j++)
        vreg[j].f4 = *(const float4*)(gV + (size_t)(j * 32) * kD);
    for (int j = 0; j < 2; j++)
        for (int e = 0; e < 8; e++)
            sVt[(c0 + e) * LD + r0 + j * 32] = vreg[j].h[e];   // transpose
    wait_async0();
    __syncthreads();

    float mrow[8], lrow[8];
    v8f O[4];
    for (int v = 0; v < 8; v++) { mrow[v] = -3.0e38f; lrow[v] = 0.0f; }
    for (int nt = 0; nt < 4; nt++)
        for (int e = 0; e < 8; e++) O[nt][e] = 0.0f;

    int buf = 0;
    for (int t = 0; t < kS; t += 64) {
        int tn = t + 64;
        int has_next = tn < kS;
        if (has_next) {
            // stream next K tile async + next V tile into registers, overlapping math
            unsigned kb = (unsigned)(buf ^ 1) * KBUFB;
            for (int j = 0; j < 2; j++)
                async_b128(lK0 + kb + (unsigned)j * ROWJ32,
                           gK + (size_t)(tn + j * 32) * kD);
            for (int j = 0; j < 2; j++)
                vreg[j].f4 = *(const float4*)(gV + (size_t)(tn + j * 32) * kD);
        }
        const _Float16* bK  = sK + buf * KVH;
        const _Float16* bVt = sVt + buf * KVH;

        // S = Q K^T * 1/sqrt(dk)  (16 rows x 64 keys per wave)
        v8f sc[4];
        for (int nt = 0; nt < 4; nt++)
            for (int e = 0; e < 8; e++) sc[nt][e] = 0.0f;
        for (int ks = 0; ks < 2; ks++) {
            v16h aq = load_frag_a(&sQ[(wave * 16) * LD + ks * 32], LD, lane);
            for (int nt = 0; nt < 4; nt++) {
                v16h bk = load_frag_b(&bK[(nt * 16) * LD + ks * 32], LD, lane);
                sc[nt] = __builtin_amdgcn_wmma_f32_16x16x32_f16(false, aq, false, bk,
                                                                (short)0, sc[nt], false, false);
            }
        }
        for (int nt = 0; nt < 4; nt++)
            for (int v = 0; v < 8; v++) sc[nt][v] *= 0.125f;

        // online softmax; row m = v + 8*(lane/16) lives in a 16-lane half-group
        float mnew[8], alpha[8], rsum[8];
        for (int v = 0; v < 8; v++) {
            float mt = fmaxf(fmaxf(sc[0][v], sc[1][v]), fmaxf(sc[2][v], sc[3][v]));
            for (int off = 1; off < 16; off <<= 1)
                mt = fmaxf(mt, __shfl_xor(mt, off, 32));
            mnew[v] = fmaxf(mrow[v], mt);
            alpha[v] = __expf(mrow[v] - mnew[v]);
            rsum[v] = 0.0f;
        }
        for (int nt = 0; nt < 4; nt++)
            for (int v = 0; v < 8; v++) {
                float p = __expf(sc[nt][v] - mnew[v]);
                sc[nt][v] = p;
                rsum[v] += p;
            }
        for (int v = 0; v < 8; v++) {
            for (int off = 1; off < 16; off <<= 1)
                rsum[v] += __shfl_xor(rsum[v], off, 32);
            lrow[v] = lrow[v] * alpha[v] + rsum[v];
            mrow[v] = mnew[v];
        }
        for (int nt = 0; nt < 4; nt++)
            for (int v = 0; v < 8; v++) O[nt][v] *= alpha[v];

        // park next V tile (transposed) into the other buffer mid-iteration
        if (has_next) {
            _Float16* dV = sVt + (buf ^ 1) * KVH;
            for (int j = 0; j < 2; j++)
                for (int e = 0; e < 8; e++)
                    dV[(c0 + e) * LD + r0 + j * 32] = vreg[j].h[e];
        }

        // C-layout -> A-layout conversion through per-wave LDS (same-wave DS is in-order)
        _Float16* pP = &sP[wave * 16 * LD];
        int mlo = (lane >> 4) << 3, nn = lane & 15;
        for (int nt = 0; nt < 4; nt++)
            for (int v = 0; v < 8; v++)
                pP[(mlo + v) * LD + nt * 16 + nn] = (_Float16)sc[nt][v];

        // O += P V  (B^T = sVt)
        for (int ks = 0; ks < 2; ks++) {
            v16h ap = load_frag_a(pP + ks * 32, LD, lane);
            for (int nt = 0; nt < 4; nt++) {
                v16h bv = load_frag_b(&bVt[(nt * 16) * LD + ks * 32], LD, lane);
                O[nt] = __builtin_amdgcn_wmma_f32_16x16x32_f16(false, ap, false, bv,
                                                               (short)0, O[nt], false, false);
            }
        }
        wait_async0();
        __syncthreads();
        buf ^= 1;
    }

    // normalize and store ctx (f16, feeds the W_o GEMM)
    int mlo = (lane >> 4) << 3, nn = lane & 15;
    for (int v = 0; v < 8; v++) {
        float inv = 1.0f / lrow[v];
        int row = q0 + wave * 16 + mlo + v;
        for (int nt = 0; nt < 4; nt++)
            ctx[(size_t)row * kD + colBase + nt * 16 + nn] = (_Float16)(O[nt][v] * inv);
    }
}

// ---------------- residual + LayerNorm ----------------
// out = LN(x + y) * g + b ; one 1024-wide row per 256-thread block
__global__ __launch_bounds__(256) void add_layernorm(const float* __restrict__ x,
                                                     const float* __restrict__ y,
                                                     const float* __restrict__ g,
                                                     const float* __restrict__ b,
                                                     float* __restrict__ outF,
                                                     _Float16* __restrict__ outH) {
    __shared__ float red[16];
    int row = blockIdx.x, tid = threadIdx.x;
    int lane = tid & 31, wave = tid >> 5;
    const float* px = x + (size_t)row * kD;
    const float* py = y + (size_t)row * kD;

    float vals[4], s = 0.0f, ss = 0.0f;
    for (int i = 0; i < 4; i++) {
        float v = px[tid + i * 256] + py[tid + i * 256];
        vals[i] = v; s += v; ss += v * v;
    }
    for (int off = 1; off < 32; off <<= 1) {
        s += __shfl_xor(s, off, 32);
        ss += __shfl_xor(ss, off, 32);
    }
    if (lane == 0) { red[wave] = s; red[8 + wave] = ss; }
    __syncthreads();
    float st = 0.0f, sst = 0.0f;
    for (int w = 0; w < 8; w++) { st += red[w]; sst += red[8 + w]; }
    float mean = st * (1.0f / kD);
    float var = sst * (1.0f / kD) - mean * mean;
    float inv = rsqrtf(var + LN_EPS);

    for (int i = 0; i < 4; i++) {
        int c = tid + i * 256;
        float v = (vals[i] - mean) * inv * g[c] + b[c];
        size_t idx = (size_t)row * kD + c;
        if (outF) outF[idx] = v;
        if (outH) outH[idx] = (_Float16)v;
    }
}

// ---------------- orchestration ----------------
extern "C" void kernel_launch(void* const* d_in, const int* in_sizes, int n_in,
                              void* d_out, int out_size, void* d_ws, size_t ws_size,
                              hipStream_t stream) {
    const float* x    = (const float*)d_in[0];
    const float* w_q  = (const float*)d_in[1];
    const float* b_q  = (const float*)d_in[2];
    const float* w_k  = (const float*)d_in[3];
    const float* b_k  = (const float*)d_in[4];
    const float* w_v  = (const float*)d_in[5];
    const float* b_v  = (const float*)d_in[6];
    const float* w_o  = (const float*)d_in[7];
    const float* b_o  = (const float*)d_in[8];
    const float* w1   = (const float*)d_in[9];
    const float* b1   = (const float*)d_in[10];
    const float* w2   = (const float*)d_in[11];
    const float* b2   = (const float*)d_in[12];
    const float* ln1g = (const float*)d_in[13];
    const float* ln1b = (const float*)d_in[14];
    const float* ln2g = (const float*)d_in[15];
    const float* ln2b = (const float*)d_in[16];
    float* out = (float*)d_out;

    // carve workspace
    char* base = (char*)d_ws;
    size_t off = 0;
    auto carve = [&](size_t bytes) -> char* {
        char* p = base + off;
        off += (bytes + 255) & ~(size_t)255;
        return p;
    };
    _Float16* x16   = (_Float16*)carve((size_t)kS * kD * 2);
    _Float16* wqT   = (_Float16*)carve((size_t)kD * kD * 2);
    _Float16* wkT   = (_Float16*)carve((size_t)kD * kD * 2);
    _Float16* wvT   = (_Float16*)carve((size_t)kD * kD * 2);
    _Float16* woT   = (_Float16*)carve((size_t)kD * kD * 2);
    _Float16* w1T   = (_Float16*)carve((size_t)kD * kDFF * 2);
    _Float16* w2T   = (_Float16*)carve((size_t)kDFF * kD * 2);
    _Float16* q16   = (_Float16*)carve((size_t)kS * kD * 2);
    _Float16* k16   = (_Float16*)carve((size_t)kS * kD * 2);
    _Float16* v16   = (_Float16*)carve((size_t)kS * kD * 2);
    _Float16* ctx16 = (_Float16*)carve((size_t)kS * kD * 2);
    float*    attnF = (float*)carve((size_t)kS * kD * 4);
    float*    y1F   = (float*)carve((size_t)kS * kD * 4);
    _Float16* y1H   = (_Float16*)carve((size_t)kS * kD * 2);
    _Float16* h1H   = (_Float16*)carve((size_t)kS * kDFF * 2);
    float*    ffF   = (float*)carve((size_t)kS * kD * 4);
    (void)ws_size; (void)in_sizes; (void)n_in; (void)out_size;

    // 1) casts
    cast_f32_to_f16<<<4096, 256, 0, stream>>>(x, x16, kS * kD);
    {
        dim3 blk(32, 8);
        transpose_cast_f16<<<dim3(kD / 32, kD / 32), blk, 0, stream>>>(w_q, wqT, kD, kD);
        transpose_cast_f16<<<dim3(kD / 32, kD / 32), blk, 0, stream>>>(w_k, wkT, kD, kD);
        transpose_cast_f16<<<dim3(kD / 32, kD / 32), blk, 0, stream>>>(w_v, wvT, kD, kD);
        transpose_cast_f16<<<dim3(kD / 32, kD / 32), blk, 0, stream>>>(w_o, woT, kD, kD);
        transpose_cast_f16<<<dim3(kDFF / 32, kD / 32), blk, 0, stream>>>(w1, w1T, kD, kDFF);
        transpose_cast_f16<<<dim3(kD / 32, kDFF / 32), blk, 0, stream>>>(w2, w2T, kDFF, kD);
    }

    // 2) QKV projections (f16 out)
    gemm_f16_wmma<<<dim3(kD / 128, kS / 128), 256, 0, stream>>>(
        x16, wqT, b_q, nullptr, q16, kS, kD, kD, 0);
    gemm_f16_wmma<<<dim3(kD / 128, kS / 128), 256, 0, stream>>>(
        x16, wkT, b_k, nullptr, k16, kS, kD, kD, 0);
    gemm_f16_wmma<<<dim3(kD / 128, kS / 128), 256, 0, stream>>>(
        x16, wvT, b_v, nullptr, v16, kS, kD, kD, 0);

    // 3) attention
    flash_attn_wmma<<<dim3(kS / 128, kH), 256, 0, stream>>>(q16, k16, v16, ctx16);

    // 4) output projection (f32 out)
    gemm_f16_wmma<<<dim3(kD / 128, kS / 128), 256, 0, stream>>>(
        ctx16, woT, b_o, attnF, nullptr, kS, kD, kD, 0);

    // 5) residual + LN1 (f32 + f16 out)
    add_layernorm<<<kS, 256, 0, stream>>>(x, attnF, ln1g, ln1b, y1F, y1H);

    // 6) FFN up + exact GELU (f16 out)
    gemm_f16_wmma<<<dim3(kDFF / 128, kS / 128), 256, 0, stream>>>(
        y1H, w1T, b1, nullptr, h1H, kS, kDFF, kD, 1);

    // 7) FFN down (f32 out)
    gemm_f16_wmma<<<dim3(kD / 128, kS / 128), 256, 0, stream>>>(
        h1H, w2T, b2, ffF, nullptr, kS, kD, kDFF, 0);

    // 8) residual + LN2 -> final output
    add_layernorm<<<kS, 256, 0, stream>>>(y1F, ffF, ln2g, ln2b, out, nullptr);
}
